// xLSTM_76982993814108
// MI455X (gfx1250) — matrix operand order repaired and checked by reading, not compiled, not run on hardware
//
#include <hip/hip_runtime.h>

// Problem constants (B,T,D,H,L) = (128,512,512,512,2)
#define BB 128
#define TT 512
#define DD 512
#define HH 512
#define LL 2
#define KC 1024   // D + H (GEMM K)
#define G4 2048   // 4*H   (gates N)
#define EPSLN 1e-5f

typedef __attribute__((ext_vector_type(16))) __bf16 bf16x16;
typedef __attribute__((ext_vector_type(8)))  __bf16 bf16x8;
typedef __attribute__((ext_vector_type(8)))  float  f32x8;

static __device__ __forceinline__ __bf16 f2bf(float f) {
  union { float f; unsigned u; } v; v.f = f;
  unsigned u = v.u;
  u += 0x7FFFu + ((u >> 16) & 1u);           // round-to-nearest-even
  unsigned short h = (unsigned short)(u >> 16);
  __bf16 r;
  __builtin_memcpy(&r, &h, 2);
  return r;
}

static __device__ __forceinline__ float wave_sum32(float v) {
#pragma unroll
  for (int off = 16; off > 0; off >>= 1) v += __shfl_xor(v, off, 32);
  return v;
}

static __device__ __forceinline__ float sigm(float x) {
  return 1.0f / (1.0f + __expf(-x));
}

// -------------------------------------------------------------------------
// Weight conversion: fp32 -> bf16 in per-lane WMMA B-fragment order.
// Fragment tile (k0,n0) of 32x16: element (k,n) -> tile base + lane*16 + e
//   lane = (n&15) + 16*((k>>4)&1),  e = k&15
// Tile index = (n>>4)*(K/32) + (k>>5); each tile = 512 bf16 (1 KB).
// -------------------------------------------------------------------------
__global__ void conv_comb_kernel(const float* __restrict__ wih,
                                 const float* __restrict__ whh,
                                 __bf16* __restrict__ dst) {
  long id = (long)blockIdx.x * blockDim.x + threadIdx.x;   // L*2048*1024 = 2^22
  int k = (int)(id & 1023);
  int n = (int)((id >> 10) & 2047);
  int l = (int)(id >> 21);
  float v = (k < DD) ? wih[((long)(l * G4 + n)) * DD + k]
                     : whh[((long)(l * G4 + n)) * HH + (k - DD)];
  long sub = ((long)(n >> 4) * 32 + (k >> 5)) * 512
           + (long)(((n & 15) + (((k >> 4) & 1) << 4)) * 16 + (k & 15));
  dst[(long)l * ((long)G4 * KC) + sub] = f2bf(v);
}

__global__ void conv_gm_kernel(const float* __restrict__ w,
                               __bf16* __restrict__ dst) {
  long id = (long)blockIdx.x * blockDim.x + threadIdx.x;   // L*512*1024 = 2^20
  int k = (int)(id & 1023);
  int n = (int)((id >> 10) & 511);
  int l = (int)(id >> 19);
  float v = w[((long)(l * HH + n)) * KC + k];
  long sub = ((long)(n >> 4) * 32 + (k >> 5)) * 512
           + (long)(((n & 15) + (((k >> 4) & 1) << 4)) * 16 + (k & 15));
  dst[(long)l * ((long)HH * KC) + sub] = f2bf(v);
}

__global__ void zero_kernel(float* __restrict__ p, long n) {
  long i = (long)blockIdx.x * blockDim.x + threadIdx.x;
  if (i < n) p[i] = 0.0f;
}

// -------------------------------------------------------------------------
// Kernel A: LayerNorm(input) + LayerNorm(h_prev) -> bf16 cat in LDS (+ global
// mirror for kernel B), then gates = cat @ [W_ih;W_hh]^T + b_ih + b_hh via
// v_wmma_f32_16x16x32_bf16.  Grid: (2048/128, 128/16) = (16, 8), 256 thr.
// -------------------------------------------------------------------------
__global__ __launch_bounds__(256) void xlstm_ln_gates(
    const float* __restrict__ inp, long inp_bstride,
    const float* __restrict__ g_in, const float* __restrict__ b_in,
    const float* __restrict__ hprev,
    const float* __restrict__ g_h, const float* __restrict__ b_h,
    const float* __restrict__ bih, const float* __restrict__ bhh,
    const __bf16* __restrict__ wfrag,
    __bf16* __restrict__ catbuf, float* __restrict__ gates) {
  __shared__ __align__(16) __bf16 shA[16][KC + 16];   // 32B row pad

  const int tid = threadIdx.x;
  const int lane = tid & 31;
  const int wv = tid >> 5;                 // 8 waves
  const int rowbase = blockIdx.y * 16;
  const int colbase = blockIdx.x * 128;

  // ---- LayerNorm: wave wv normalizes rows 2wv, 2wv+1 (x part + h part) ----
#pragma unroll
  for (int rr = 0; rr < 2; ++rr) {
    const int r = wv * 2 + rr;
    const int b = rowbase + r;
    // x part (D = 512)
    {
      const float* src = inp + (long)b * inp_bstride;
      float xs[16];
      float s = 0.f, ss = 0.f;
#pragma unroll
      for (int i = 0; i < 16; ++i) {
        float v = src[lane + 32 * i];
        xs[i] = v; s += v; ss += v * v;
      }
      s = wave_sum32(s); ss = wave_sum32(ss);
      float mu = s * (1.0f / DD);
      float rstd = rsqrtf(ss * (1.0f / DD) - mu * mu + EPSLN);
#pragma unroll
      for (int i = 0; i < 16; ++i) {
        int d = lane + 32 * i;
        __bf16 bv = f2bf((xs[i] - mu) * rstd * g_in[d] + b_in[d]);
        shA[r][d] = bv;
        if (blockIdx.x == 0) catbuf[(long)b * KC + d] = bv;
      }
    }
    // h part (H = 512)
    {
      const float* src = hprev + (long)b * HH;
      float xs[16];
      float s = 0.f, ss = 0.f;
#pragma unroll
      for (int i = 0; i < 16; ++i) {
        float v = src[lane + 32 * i];
        xs[i] = v; s += v; ss += v * v;
      }
      s = wave_sum32(s); ss = wave_sum32(ss);
      float mu = s * (1.0f / HH);
      float rstd = rsqrtf(ss * (1.0f / HH) - mu * mu + EPSLN);
#pragma unroll
      for (int i = 0; i < 16; ++i) {
        int d = lane + 32 * i;
        __bf16 bv = f2bf((xs[i] - mu) * rstd * g_h[d] + b_h[d]);
        shA[r][DD + d] = bv;
        if (blockIdx.x == 0) catbuf[(long)b * KC + DD + d] = bv;
      }
    }
  }
  __syncthreads();

  // ---- WMMA GEMM: 16x16 tile per wave, K = 1024 (32 steps of K=32) ----
  const int ntile = blockIdx.x * 8 + wv;           // 0..127
  const int col = colbase + wv * 16 + (lane & 15);
  const float bias = bih[col] + bhh[col];
  f32x8 acc;
#pragma unroll
  for (int i = 0; i < 8; ++i) acc[i] = bias;

  const int arow = lane & 15;
  const int khalf = (lane >> 4) ? 16 : 0;
  const __bf16* wp = wfrag + (long)ntile * 32 * 512 + lane * 16;

#pragma unroll 4
  for (int kt = 0; kt < 32; ++kt) {
    const int k0 = kt * 32;
    bf16x8 alo = *(const bf16x8*)&shA[arow][k0 + khalf];
    bf16x8 ahi = *(const bf16x8*)&shA[arow][k0 + 8 + khalf];
    bf16x16 a;
#pragma unroll
    for (int i = 0; i < 8; ++i) { a[i] = alo[i]; a[8 + i] = ahi[i]; }
    bf16x16 bfr = *(const bf16x16*)(wp + (long)kt * 512);
    if (kt + 1 < 32) __builtin_prefetch((const void*)(wp + (long)(kt + 1) * 512), 0, 3);
    acc = __builtin_amdgcn_wmma_f32_16x16x32_bf16(false, a, false, bfr,
                                                  (short)0, acc, false, false);
  }

  const int r0 = rowbase + ((lane >> 4) ? 8 : 0);
#pragma unroll
  for (int v = 0; v < 8; ++v)
    gates[(long)(r0 + v) * G4 + col] = acc[v];
}

// -------------------------------------------------------------------------
// Kernel B: g = sigmoid(cat@Wg^T+bg), m = sigmoid(cat@Wm^T+bm) via WMMA
// (shared A fragment), then all pointwise LSTM/state updates.
// Grid: (512/128, 128/16) = (4, 8), 256 threads.
// -------------------------------------------------------------------------
__global__ __launch_bounds__(256) void xlstm_gm_update(
    const __bf16* __restrict__ catbuf, const float* __restrict__ gates,
    const __bf16* __restrict__ wgfrag, const __bf16* __restrict__ wmfrag,
    const float* __restrict__ bg, const float* __restrict__ bmv,
    float* __restrict__ h, float* __restrict__ c,
    float* __restrict__ out) {
  __shared__ __align__(16) __bf16 shA[16][KC + 16];

  const int tid = threadIdx.x;
  const int lane = tid & 31;
  const int wv = tid >> 5;
  const int rowbase = blockIdx.y * 16;
  const int colbase = blockIdx.x * 128;

  // cooperative 32 KB copy of 16 cat rows into LDS (16B vectors)
#pragma unroll
  for (int i = 0; i < 8; ++i) {
    int chunk = tid + 256 * i;          // 2048 chunks of 8 bf16
    int r = chunk >> 7;
    int off = (chunk & 127) * 8;
    uint4 val = *(const uint4*)(catbuf + (long)(rowbase + r) * KC + off);
    *(uint4*)&shA[r][off] = val;
  }
  __syncthreads();

  const int ntile = blockIdx.x * 8 + wv;           // 0..31 over H
  const int col = colbase + wv * 16 + (lane & 15);
  f32x8 accG, accM;
  {
    float g0 = bg[col], m0 = bmv[col];
#pragma unroll
    for (int i = 0; i < 8; ++i) { accG[i] = g0; accM[i] = m0; }
  }

  const int arow = lane & 15;
  const int khalf = (lane >> 4) ? 16 : 0;
  const __bf16* wpg = wgfrag + (long)ntile * 32 * 512 + lane * 16;
  const __bf16* wpm = wmfrag + (long)ntile * 32 * 512 + lane * 16;

#pragma unroll 2
  for (int kt = 0; kt < 32; ++kt) {
    const int k0 = kt * 32;
    bf16x8 alo = *(const bf16x8*)&shA[arow][k0 + khalf];
    bf16x8 ahi = *(const bf16x8*)&shA[arow][k0 + 8 + khalf];
    bf16x16 a;
#pragma unroll
    for (int i = 0; i < 8; ++i) { a[i] = alo[i]; a[8 + i] = ahi[i]; }
    bf16x16 bgf = *(const bf16x16*)(wpg + (long)kt * 512);
    bf16x16 bmf = *(const bf16x16*)(wpm + (long)kt * 512);
    accG = __builtin_amdgcn_wmma_f32_16x16x32_bf16(false, a, false, bgf,
                                                   (short)0, accG, false, false);
    accM = __builtin_amdgcn_wmma_f32_16x16x32_bf16(false, a, false, bmf,
                                                   (short)0, accM, false, false);
  }

  // pointwise LSTM + mixing-gate state update
  const int r0 = rowbase + ((lane >> 4) ? 8 : 0);
#pragma unroll
  for (int v = 0; v < 8; ++v) {
    const int b = r0 + v;
    const long gb = (long)b * G4 + col;
    const long idx = (long)b * HH + col;
    float ig = sigm(gates[gb]);
    float fg = sigm(gates[gb + HH]);
    float gg = tanhf(gates[gb + 2 * HH]);
    float og = sigm(gates[gb + 3 * HH]);
    float cp = c[idx];
    float hp = h[idx];
    float cn = fg * cp + ig * gg;
    float tc = tanhf(cn);
    float hnew = og * tc;
    float gs = sigm(accG[v]);
    float ms = sigm(accM[v]);
    float ho = gs * hnew + (1.0f - gs) * hp;
    float co = ms * cn + (1.0f - ms) * tc;
    h[idx] = ho;
    c[idx] = co;
    if (out) out[idx] = ho;
  }
}

// -------------------------------------------------------------------------
// Host side
// -------------------------------------------------------------------------
extern "C" void kernel_launch(void* const* d_in, const int* in_sizes, int n_in,
                              void* d_out, int out_size, void* d_ws, size_t ws_size,
                              hipStream_t stream) {
  (void)in_sizes; (void)n_in; (void)out_size; (void)ws_size;
  const float* x    = (const float*)d_in[0];
  const float* lnig = (const float*)d_in[1];
  const float* lnib = (const float*)d_in[2];
  const float* lnhg = (const float*)d_in[3];
  const float* lnhb = (const float*)d_in[4];
  const float* Wih  = (const float*)d_in[5];
  const float* Whh  = (const float*)d_in[6];
  const float* bih  = (const float*)d_in[7];
  const float* bhh  = (const float*)d_in[8];
  const float* Wg   = (const float*)d_in[9];
  const float* bg   = (const float*)d_in[10];
  const float* Wm   = (const float*)d_in[11];
  const float* bm   = (const float*)d_in[12];

  char* ws = (char*)d_ws;
  __bf16* wcomb = (__bf16*)(ws);                                   // 8 MiB
  __bf16* wgf   = (__bf16*)(ws + (8ul << 20));                     // 2 MiB
  __bf16* wmf   = (__bf16*)(ws + (10ul << 20));                    // 2 MiB
  __bf16* cat   = (__bf16*)(ws + (12ul << 20));                    // 256 KiB
  float*  gates = (float*)(ws + (12ul << 20) + (256ul << 10));     // 1 MiB
  float*  hbuf  = (float*)(ws + (12ul << 20) + (1280ul << 10));    // 512 KiB
  float*  cbuf  = (float*)(ws + (12ul << 20) + (1792ul << 10));    // 512 KiB

  // One-time (per launch) weight conversion into WMMA-fragment bf16 layout.
  conv_comb_kernel<<<(LL * G4 * KC) / 256, 256, 0, stream>>>(Wih, Whh, wcomb);
  conv_gm_kernel<<<(LL * HH * KC) / 256, 256, 0, stream>>>(Wg, wgf);
  conv_gm_kernel<<<(LL * HH * KC) / 256, 256, 0, stream>>>(Wm, wmf);
  // Zero h0/c0 (hbuf and cbuf are contiguous: 2 * L*B*H floats).
  zero_kernel<<<(2l * LL * BB * HH + 255) / 256, 256, 0, stream>>>(
      hbuf, 2l * LL * BB * HH);

  for (int t = 0; t < TT; ++t) {
    for (int l = 0; l < LL; ++l) {
      const float* inp = (l == 0) ? (x + (long)t * DD) : hbuf;
      long stride = (l == 0) ? (long)TT * DD : (long)HH;
      float* hl = hbuf + (long)l * BB * HH;
      float* cl = cbuf + (long)l * BB * HH;
      xlstm_ln_gates<<<dim3(16, 8), 256, 0, stream>>>(
          inp, stride, lnig + l * DD, lnib + l * DD, hl,
          lnhg + l * HH, lnhb + l * HH, bih + l * G4, bhh + l * G4,
          wcomb + (long)l * G4 * KC, cat, gates);
      float* outp = (t == TT - 1 && l == LL - 1) ? (float*)d_out : nullptr;
      xlstm_gm_update<<<dim3(4, 8), 256, 0, stream>>>(
          cat, gates, wgf + (long)l * HH * KC, wmf + (long)l * HH * KC,
          bg + l * HH, bm + l * HH, hl, cl, outp);
    }
  }
}
